// GATEncoder_36438502539674
// MI455X (gfx1250) — matrix operand order repaired and checked by reading, not compiled, not run on hardware
//
#include <hip/hip_runtime.h>
#include <hip/hip_bf16.h>
#include <math.h>

// ---------------------------------------------------------------------------
// GATConv forward (single head) for MI455X / gfx1250 (wave32, WMMA + TDM).
//   h = x @ W                         (WMMA f16->f32, 16x16x32 tiles)
//   a_s = h@att_src, a_d = h@att_dst
//   e   = leaky_relu(a_s[src]+a_d[dst], 0.2)   over E + N self-loop edges
//   alpha = softmax over incoming edges of dst (stable, segment max)
//   out = elu(segment_sum(alpha * h[src]) + bias)
// Outputs (concatenated, float): out[N*48], full_edge_index[2*Etot], alpha[Etot]
// ---------------------------------------------------------------------------

#define IN_F  128
#define OUT_F 48
#define NEG_SLOPE 0.2f
#define WT_LD 136                 // padded LDS stride (halves): 272B = 68 banks

typedef _Float16 v16h __attribute__((ext_vector_type(16)));
typedef _Float16 v8h  __attribute__((ext_vector_type(8)));
typedef float    v8f  __attribute__((ext_vector_type(8)));
typedef unsigned int u32x4 __attribute__((ext_vector_type(4)));
typedef int          i32x4 __attribute__((ext_vector_type(4)));
typedef int          i32x8 __attribute__((ext_vector_type(8)));

#if defined(__HIP_DEVICE_COMPILE__) && __has_builtin(__builtin_amdgcn_tensor_load_to_lds)
#define HAVE_TDM 1
#else
#define HAVE_TDM 0
#endif

// order-preserving float <-> uint encoding for atomicMax on signed floats
__device__ __forceinline__ unsigned enc_f32(float f) {
    unsigned u = __float_as_uint(f);
    return (u & 0x80000000u) ? ~u : (u | 0x80000000u);
}
__device__ __forceinline__ float dec_f32(unsigned u) {
    unsigned v = (u & 0x80000000u) ? (u & 0x7FFFFFFFu) : ~u;
    return __uint_as_float(v);
}

// ---------------------------------------------------------------------------
// Kernel 1: h = x @ W via v_wmma_f32_16x16x32_f16.
// One wave -> one 16-row M tile x all 48 columns (3 16x16 accumulators).
// 8 waves / 256-thread block -> 128 rows per block.
// W is staged into LDS by the Tensor Data Mover (2D tile DMA), then converted
// to f16 *transposed* (WT[n][k], padded stride) so each lane's B fragment is
// 16 contiguous halves -> 2x ds_load_b128 instead of 16x ds_load_u16.
// Fragment packing used here (wave32, contiguous k, same for A and B):
//   A 16x32 f16 : lanes 0-15 row M=lane, elems 0..15 = K+0..15
//                 lanes16-31 row M=lane-16, elems 0..15 = K+16..31
//   B 32x16 f16 : lane holds column n0+(lane&15); elems t = K + 16*(lane>>4) + t
//   D 16x16 f32 : vgpr r, lanes0-15 = D[r][lane], lanes16-31 = D[8+r][lane-16]
// ---------------------------------------------------------------------------
__global__ void __launch_bounds__(256)
gat_gemm_wmma(const float* __restrict__ x, const float* __restrict__ W,
              float* __restrict__ h, int N, int nTiles) {
    __shared__ _Float16 WT[OUT_F * WT_LD];           // transposed f16 W, ~12.8 KB

#if HAVE_TDM
    __shared__ float Wstage[IN_F * OUT_F];           // 24 KB f32 staging for TDM
    if (threadIdx.x == 0) {
        // ---- Tensor DMA descriptor (ISA 8.3/8.4): one 128x48 f32 2D tile ----
        unsigned long long ga = (unsigned long long)(uintptr_t)W;
        unsigned lds_off = (unsigned)(uintptr_t)(void*)Wstage;   // addr[31:0] = LDS offset
        u32x4 g0;
        g0[0] = 1u;                                              // count=1, user desc
        g0[1] = lds_off;                                         // lds_addr
        g0[2] = (unsigned)(ga & 0xFFFFFFFFull);                  // global_addr[31:0]
        g0[3] = (unsigned)((ga >> 32) & 0x1FFFFFFull) | (2u << 30); // addr[56:32] | type=2
        i32x8 g1;
        g1[0] = (int)(2u << 16);                 // workgroup_mask=0, data_size=4B
        g1[1] = (int)((unsigned)OUT_F << 16);    // tensor_dim0=48 (low 16b @ bit48)
        g1[2] = (int)((unsigned)IN_F  << 16);    // tensor_dim0 hi=0 | tensor_dim1=128
        g1[3] = (int)((unsigned)OUT_F << 16);    // tensor_dim1 hi=0 | tile_dim0=48
        g1[4] = (int)IN_F;                       // tile_dim1=128, tile_dim2=0
        g1[5] = (int)OUT_F;                      // tensor_dim0_stride=48 (low 32b)
        g1[6] = 0;                               // stride hi | tensor_dim1_stride lo
        g1[7] = 0;
        i32x4 z4 = {0, 0, 0, 0};
#if __has_include(<hip/amd_detail/amd_gfx1250_TDM.h>)
        i32x8 z8 = {0, 0, 0, 0, 0, 0, 0, 0};
        __builtin_amdgcn_tensor_load_to_lds(g0, g1, z4, z4, z8, 0);   // 6-arg toolchain
#else
        __builtin_amdgcn_tensor_load_to_lds(g0, g1, z4, z4, 0);       // 5-arg toolchain
#endif
        __builtin_amdgcn_s_wait_tensorcnt(0);
    }
    __syncthreads();
    const float* Wsrc = Wstage;
#else
    const float* Wsrc = W;
#endif

    // convert + transpose into WT[n][k]
    for (int i = threadIdx.x; i < IN_F * OUT_F; i += 256) {
        int k = i / OUT_F, n = i - k * OUT_F;
        WT[n * WT_LD + k] = (_Float16)Wsrc[i];
    }
    __syncthreads();

    const int wave = threadIdx.x >> 5;
    const int lane = threadIdx.x & 31;
    const int tile = blockIdx.x * 8 + wave;
    if (tile >= nTiles) return;                      // wave-uniform: EXEC stays full

    const int hl  = lane >> 4;                       // which 16-lane half
    const int l16 = lane & 15;
    int row = tile * 16 + l16;                       // row this lane supplies for A
    if (row >= N) row = N - 1;                       // clamp (padding rows unused)

    v8f acc0 = {}, acc1 = {}, acc2 = {};

    for (int k0 = 0; k0 < IN_F; k0 += 32) {
        // ---- A fragment (global, fp32 -> f16) ----
        const float* xa = x + (size_t)row * IN_F + k0 + hl * 16;  // 32B aligned
        const float4* xv = reinterpret_cast<const float4*>(xa);
        float4 p0 = xv[0], p1 = xv[1];               // K offsets +0..7  (of half)
        float4 q0 = xv[2], q1 = xv[3];               // K offsets +8..15
        v16h a;
        a[0]=(_Float16)p0.x; a[1]=(_Float16)p0.y; a[2]=(_Float16)p0.z; a[3]=(_Float16)p0.w;
        a[4]=(_Float16)p1.x; a[5]=(_Float16)p1.y; a[6]=(_Float16)p1.z; a[7]=(_Float16)p1.w;
        a[8]=(_Float16)q0.x; a[9]=(_Float16)q0.y; a[10]=(_Float16)q0.z; a[11]=(_Float16)q0.w;
        a[12]=(_Float16)q1.x; a[13]=(_Float16)q1.y; a[14]=(_Float16)q1.z; a[15]=(_Float16)q1.w;

        // ---- B fragments from transposed LDS: 16 contiguous halves per lane ----
        const int kb = k0 + hl * 16;
        const _Float16* c0 = &WT[(size_t)(l16 +  0) * WT_LD + kb];
        const _Float16* c1 = &WT[(size_t)(l16 + 16) * WT_LD + kb];
        const _Float16* c2 = &WT[(size_t)(l16 + 32) * WT_LD + kb];
        v8h b0l = *(const v8h*)c0,       b0h = *(const v8h*)(c0 + 8);
        v8h b1l = *(const v8h*)c1,       b1h = *(const v8h*)(c1 + 8);
        v8h b2l = *(const v8h*)c2,       b2h = *(const v8h*)(c2 + 8);
        v16h b0 = __builtin_shufflevector(b0l, b0h, 0,1,2,3,4,5,6,7,8,9,10,11,12,13,14,15);
        v16h b1 = __builtin_shufflevector(b1l, b1h, 0,1,2,3,4,5,6,7,8,9,10,11,12,13,14,15);
        v16h b2 = __builtin_shufflevector(b2l, b2h, 0,1,2,3,4,5,6,7,8,9,10,11,12,13,14,15);

        acc0 = __builtin_amdgcn_wmma_f32_16x16x32_f16(false, a, false, b0,
                                                      (short)0, acc0, false, false);
        acc1 = __builtin_amdgcn_wmma_f32_16x16x32_f16(false, a, false, b1,
                                                      (short)0, acc1, false, false);
        acc2 = __builtin_amdgcn_wmma_f32_16x16x32_f16(false, a, false, b2,
                                                      (short)0, acc2, false, false);
    }

    // ---- store D tiles (full-tile fast path is wave-uniform) ----
    const int rbase = tile * 16 + hl * 8;
    if (tile * 16 + 16 <= N) {
        #pragma unroll
        for (int r = 0; r < 8; ++r) {
            float* hr = h + (size_t)(rbase + r) * OUT_F + l16;
            hr[0]  = acc0[r];
            hr[16] = acc1[r];
            hr[32] = acc2[r];
        }
    } else {
        #pragma unroll
        for (int r = 0; r < 8; ++r) {
            if (rbase + r < N) {
                float* hr = h + (size_t)(rbase + r) * OUT_F + l16;
                hr[0]  = acc0[r];
                hr[16] = acc1[r];
                hr[32] = acc2[r];
            }
        }
    }
}

// ---------------------------------------------------------------------------
// Kernel 2: per-node projections + state init.
// ---------------------------------------------------------------------------
__global__ void __launch_bounds__(256)
gat_node_init(const float* __restrict__ h,
              const float* __restrict__ att_src, const float* __restrict__ att_dst,
              float* __restrict__ a_s, float* __restrict__ a_d,
              unsigned* __restrict__ m_enc, float* __restrict__ denom,
              float* __restrict__ out, int N) {
    int i = blockIdx.x * blockDim.x + threadIdx.x;
    if (i >= N) return;
    const float* hr = h + (size_t)i * OUT_F;
    float s = 0.f, d = 0.f;
    #pragma unroll
    for (int c = 0; c < OUT_F; ++c) {
        float v = hr[c];
        s += v * att_src[c];
        d += v * att_dst[c];
    }
    a_s[i] = s;
    a_d[i] = d;
    m_enc[i] = enc_f32(-INFINITY);
    denom[i] = 0.f;
    float* orow = out + (size_t)i * OUT_F;
    #pragma unroll
    for (int c = 0; c < OUT_F; ++c) orow[c] = 0.f;
}

// ---------------------------------------------------------------------------
// Kernel 3: e = leaky_relu(a_s[src]+a_d[dst]); segment max via atomicMax(enc).
// Also emits full_edge_index (as float, exact up to 2^24).
// ---------------------------------------------------------------------------
__global__ void __launch_bounds__(256)
gat_edge_max(const int* __restrict__ ei,
             const float* __restrict__ a_s, const float* __restrict__ a_d,
             float* __restrict__ e_buf, unsigned* __restrict__ m_enc,
             float* __restrict__ fei, int E, int Etot) {
    int idx = blockIdx.x * blockDim.x + threadIdx.x;
    if (idx >= Etot) return;
    int s, d;
    if (idx < E) { s = ei[idx]; d = ei[E + idx]; }
    else         { s = d = idx - E; }                 // self loops
    float t = a_s[s] + a_d[d];
    float e = t > 0.f ? t : NEG_SLOPE * t;
    e_buf[idx] = e;
    atomicMax(&m_enc[d], enc_f32(e));
    fei[idx]        = (float)s;                       // row 0: src
    fei[Etot + idx] = (float)d;                       // row 1: dst
}

// ---------------------------------------------------------------------------
// Kernel 4: ex = exp(e - m[dst]); segment sum via atomicAdd.
// ---------------------------------------------------------------------------
__global__ void __launch_bounds__(256)
gat_edge_exp(const int* __restrict__ ei,
             float* __restrict__ e_buf, const unsigned* __restrict__ m_enc,
             float* __restrict__ denom, int E, int Etot) {
    int idx = blockIdx.x * blockDim.x + threadIdx.x;
    if (idx >= Etot) return;
    int d = (idx < E) ? ei[E + idx] : (idx - E);
    float m  = dec_f32(m_enc[d]);
    float ex = __expf(e_buf[idx] - m);
    e_buf[idx] = ex;                                  // overwrite e with ex
    atomicAdd(&denom[d], ex);
}

// ---------------------------------------------------------------------------
// Kernel 5: alpha = ex/denom[dst]; out[dst,:] += alpha * h[src,:].
// 16 lanes per edge, 3 columns per lane (48 total); lane 0 writes alpha.
// ---------------------------------------------------------------------------
__global__ void __launch_bounds__(256)
gat_edge_scatter(const int* __restrict__ ei, const float* __restrict__ h,
                 const float* __restrict__ e_buf, const float* __restrict__ denom,
                 float* __restrict__ out, float* __restrict__ alpha_out,
                 int E, int Etot) {
    int gid = blockIdx.x * blockDim.x + threadIdx.x;
    int idx = gid >> 4;
    int t   = gid & 15;
    if (idx >= Etot) return;
    int s, d;
    if (idx < E) { s = ei[idx]; d = ei[E + idx]; }
    else         { s = d = idx - E; }
    float alpha = e_buf[idx] / denom[d];
    if (t == 0) alpha_out[idx] = alpha;
    const float* hs = h + (size_t)s * OUT_F;
    float* od = out + (size_t)d * OUT_F;
    atomicAdd(&od[t],      alpha * hs[t]);
    atomicAdd(&od[t + 16], alpha * hs[t + 16]);
    atomicAdd(&od[t + 32], alpha * hs[t + 32]);
}

// ---------------------------------------------------------------------------
// Kernel 6: out = elu(out + bias)
// ---------------------------------------------------------------------------
__global__ void __launch_bounds__(256)
gat_finalize(float* __restrict__ out, const float* __restrict__ bias, int total) {
    int i = blockIdx.x * blockDim.x + threadIdx.x;
    if (i >= total) return;
    float v = out[i] + bias[i % OUT_F];
    out[i] = v > 0.f ? v : (__expf(v) - 1.f);
}

// ---------------------------------------------------------------------------
extern "C" void kernel_launch(void* const* d_in, const int* in_sizes, int n_in,
                              void* d_out, int out_size, void* d_ws, size_t ws_size,
                              hipStream_t stream) {
    const float* x       = (const float*)d_in[0];
    const int*   ei      = (const int*)  d_in[1];   // [2*E], row0=src, row1=dst
    const float* W       = (const float*)d_in[2];
    const float* att_src = (const float*)d_in[3];
    const float* att_dst = (const float*)d_in[4];
    const float* bias    = (const float*)d_in[5];

    const int N    = in_sizes[0] / IN_F;
    const int E    = in_sizes[1] / 2;
    const int Etot = E + N;                          // with self loops

    // workspace layout (all fp32/u32, 4B aligned)
    char* ws = (char*)d_ws;
    float*    h      = (float*)ws;                        ws += (size_t)N * OUT_F * 4;
    float*    a_s    = (float*)ws;                        ws += (size_t)N * 4;
    float*    a_d    = (float*)ws;                        ws += (size_t)N * 4;
    unsigned* m_enc  = (unsigned*)ws;                     ws += (size_t)N * 4;
    float*    denom  = (float*)ws;                        ws += (size_t)N * 4;
    float*    e_buf  = (float*)ws;                        /* Etot * 4 */

    // output layout: out [N*48] | full_edge_index [2*Etot] | alpha [Etot]
    float* out       = (float*)d_out;
    float* fei       = out + (size_t)N * OUT_F;
    float* alpha_out = fei + (size_t)2 * Etot;

    const int nTiles = (N + 15) / 16;

    // 1) h = x @ W (TDM-staged W, WMMA)
    gat_gemm_wmma<<<(nTiles + 7) / 8, 256, 0, stream>>>(x, W, h, N, nTiles);

    // 2) projections + init
    gat_node_init<<<(N + 255) / 256, 256, 0, stream>>>(h, att_src, att_dst,
                                                       a_s, a_d, m_enc, denom, out, N);

    // 3) edge scores + segment max
    gat_edge_max<<<(Etot + 255) / 256, 256, 0, stream>>>(ei, a_s, a_d, e_buf, m_enc,
                                                         fei, E, Etot);

    // 4) exp + segment sum
    gat_edge_exp<<<(Etot + 255) / 256, 256, 0, stream>>>(ei, e_buf, m_enc, denom, E, Etot);

    // 5) alpha + weighted scatter (16 lanes / edge)
    {
        long long threads = (long long)Etot * 16;
        int blocks = (int)((threads + 255) / 256);
        gat_edge_scatter<<<blocks, 256, 0, stream>>>(ei, h, e_buf, denom, out,
                                                     alpha_out, E, Etot);
    }

    // 6) bias + ELU
    {
        int total = N * OUT_F;
        gat_finalize<<<(total + 255) / 256, 256, 0, stream>>>(out, bias, total);
    }
}
